// DeltaNet_34668976013857
// MI455X (gfx1250) — matrix operand order repaired and checked by reading, not compiled, not run on hardware
//
#include <hip/hip_runtime.h>

#define DIM   1024
#define NS    3072        // k|v|q columns of s (lr column handled separately)
#define BATCH 16384

typedef __attribute__((ext_vector_type(16))) __bf16 v16bf;
typedef __attribute__((ext_vector_type(8)))  float  v8f;
typedef __attribute__((ext_vector_type(4)))  int    v4i;

union BF16x16 { unsigned int u[8]; v16bf v; };
union U16x8   { uint4 q; unsigned short h[8]; };

typedef __attribute__((address_space(1))) v4i gv4i;   // global <4 x i32>
typedef __attribute__((address_space(3))) v4i lv4i;   // LDS    <4 x i32>

#if __has_builtin(__builtin_amdgcn_global_load_async_to_lds_b128)
#define HAVE_ASYNC_LDS 1
#else
#define HAVE_ASYNC_LDS 0
#endif

// copy 16 bytes global -> LDS; async path writes LDS directly (ASYNCcnt),
// fallback goes through VGPRs (LOADcnt + DScnt)
__device__ __forceinline__ void copy16_to_lds(const unsigned short* g,
                                              unsigned short* l) {
#if HAVE_ASYNC_LDS
  __builtin_amdgcn_global_load_async_to_lds_b128(
      (gv4i*)(void*)g, (lv4i*)l, 0, 0);
#else
  *(uint4*)l = *(const uint4*)g;
#endif
}

__device__ __forceinline__ void wait_async0() {
#if HAVE_ASYNC_LDS
#if __has_builtin(__builtin_amdgcn_s_wait_asynccnt)
  __builtin_amdgcn_s_wait_asynccnt(0);
#else
  asm volatile("s_wait_asynccnt 0x0" ::: "memory");
#endif
#endif
}

__device__ __forceinline__ unsigned short f2bf(float f) {
  unsigned int u = __float_as_uint(f);
  u += 0x7FFFu + ((u >> 16) & 1u);     // round-to-nearest-even
  return (unsigned short)(u >> 16);
}

// ---------------------------------------------------------------- converters
__global__ void k_f32_to_bf16(const float* __restrict__ in,
                              unsigned short* __restrict__ out, long n) {
  long i = (long)blockIdx.x * blockDim.x + threadIdx.x;
  long stride = (long)gridDim.x * blockDim.x;
  for (; i < n; i += stride) out[i] = f2bf(in[i]);
}

// ---------------------------------------------------------------- WMMA GEMM
// C[M x N] = A(bf16, row-major, lda) @ W^T (W bf16, N x K row-major, ldw) + bias
// mode 0 : outF[m*ldo+n] = acc + bias[n]
// mode 1 : outB[m*DIM+n] = bf16( lrv[m] * (vsrc[m*ldv+n] - (acc+bias[n])) )
#define TM 128
#define TN 128
#define TK 32
#define LSTR 40   // LDS row stride (elements), multiple of 8 -> 16B-aligned frags

__global__ __launch_bounds__(256) void k_wmma_gemm(
    const unsigned short* __restrict__ A, int lda,
    const unsigned short* __restrict__ W, int ldw,
    const float* __restrict__ bias, int K, int mode,
    float* __restrict__ outF, int ldo,
    unsigned short* __restrict__ outB,
    const float* __restrict__ lrv,
    const float* __restrict__ vsrc, int ldv)
{
  __shared__ unsigned short lA[2][TM * LSTR];   // double-buffered
  __shared__ unsigned short lB[2][TN * LSTR];

  const int tid  = threadIdx.x;
  const int wave = tid >> 5;
  const int lane = tid & 31;
  const int half = lane >> 4;
  const int l16  = lane & 15;

  const long mBase = (long)blockIdx.y * TM;
  const long nBase = (long)blockIdx.x * TN;

  v8f acc[8] = {};

  const int ldRow = tid >> 1;          // 0..127
  const int ldCol = (tid & 1) * 16;    // 0 / 16

  // stage one 128x32 A tile + 128x32 W tile into LDS buffer `b`
  auto stage = [&](int b, int k0) {
    const unsigned short* srcA = A + (mBase + ldRow) * lda + k0 + ldCol;
    const unsigned short* srcW = W + (nBase + ldRow) * ldw + k0 + ldCol;
    unsigned short* dA = &lA[b][ldRow * LSTR + ldCol];
    unsigned short* dB = &lB[b][ldRow * LSTR + ldCol];
    copy16_to_lds(srcA,     dA);
    copy16_to_lds(srcA + 8, dA + 8);
    copy16_to_lds(srcW,     dB);
    copy16_to_lds(srcW + 8, dB + 8);
  };

  stage(0, 0);
  int buf = 0;
  for (int k0 = 0; k0 < K; k0 += TK, buf ^= 1) {
    wait_async0();        // our async writes into lX[buf] are done
    __syncthreads();      // everyone's are done; prior reads of buf^1 retired
    if (k0 + TK < K) stage(buf ^ 1, k0 + TK);   // background copy of next tile

    // B fragment: lane = column (l16) of this wave's 16-col stripe,
    // K = half*8 .. +7, then 16+half*8 .. +7 (same order as A)
    BF16x16 bf;
    {
      const unsigned short* p = &lB[buf][(wave * 16 + l16) * LSTR + half * 8];
      uint4 q0 = *(const uint4*)(p);
      uint4 q1 = *(const uint4*)(p + 16);
      bf.u[0]=q0.x; bf.u[1]=q0.y; bf.u[2]=q0.z; bf.u[3]=q0.w;
      bf.u[4]=q1.x; bf.u[5]=q1.y; bf.u[6]=q1.z; bf.u[7]=q1.w;
    }
    // batch-load ALL A fragments first so ds_loads clause together and the
    // 8 WMMAs issue back-to-back
    BF16x16 af[8];
#pragma unroll
    for (int mt = 0; mt < 8; ++mt) {
      const unsigned short* p = &lA[buf][(mt * 16 + l16) * LSTR + half * 8];
      uint4 q0 = *(const uint4*)(p);        // K = half*8 .. +7
      uint4 q1 = *(const uint4*)(p + 16);   // K = 16+half*8 .. +7
      af[mt].u[0]=q0.x; af[mt].u[1]=q0.y; af[mt].u[2]=q0.z; af[mt].u[3]=q0.w;
      af[mt].u[4]=q1.x; af[mt].u[5]=q1.y; af[mt].u[6]=q1.z; af[mt].u[7]=q1.w;
    }
#pragma unroll
    for (int mt = 0; mt < 8; ++mt)
      acc[mt] = __builtin_amdgcn_wmma_f32_16x16x32_bf16(
          false, af[mt].v, false, bf.v, (short)0, acc[mt], false, false);
  }

  // D layout: VGPR i -> M = i (lanes 0-15) / 8+i (lanes 16-31), N = l16
  const long nCol = nBase + wave * 16 + l16;
  const float bn = bias[nCol];
#pragma unroll
  for (int mt = 0; mt < 8; ++mt) {
#pragma unroll
    for (int i = 0; i < 8; ++i) {
      long m = mBase + mt * 16 + half * 8 + i;
      float val = acc[mt][i] + bn;
      if (mode == 0) {
        outF[m * ldo + nCol] = val;
      } else {
        float rr = lrv[m] * (vsrc[m * ldv + nCol] - val);
        outB[m * (long)DIM + nCol] = f2bf(rr);
      }
    }
  }
}

// ---------------------------------------------------------------- lr column
__global__ __launch_bounds__(256) void k_lr(const float* __restrict__ x,
                                            const float* __restrict__ wrow,
                                            const float* __restrict__ brow,
                                            float* __restrict__ lr) {
  __shared__ float red[256];
  long row = blockIdx.x;
  float s = 0.f;
  for (int i = threadIdx.x; i < DIM; i += 256)
    s += x[row * DIM + i] * wrow[i];
  red[threadIdx.x] = s;
  __syncthreads();
  for (int off = 128; off > 0; off >>= 1) {
    if (threadIdx.x < off) red[threadIdx.x] += red[threadIdx.x + off];
    __syncthreads();
  }
  if (threadIdx.x == 0) lr[row] = 1.f / (1.f + __expf(-(red[0] + brow[0])));
}

// ---------------------------------------------------------------- activations
__device__ __forceinline__ float blockMax(float v, float* red) {
  int t = threadIdx.x;
  red[t] = v; __syncthreads();
  for (int off = 128; off > 0; off >>= 1) {
    if (t < off) red[t] = fmaxf(red[t], red[t + off]);
    __syncthreads();
  }
  float r = red[0]; __syncthreads();
  return r;
}
__device__ __forceinline__ float blockSum(float v, float* red) {
  int t = threadIdx.x;
  red[t] = v; __syncthreads();
  for (int off = 128; off > 0; off >>= 1) {
    if (t < off) red[t] += red[t + off];
    __syncthreads();
  }
  float r = red[0]; __syncthreads();
  return r;
}

__global__ __launch_bounds__(256) void k_act(const float* __restrict__ s,
                                             unsigned short* __restrict__ sk,
                                             unsigned short* __restrict__ gk,
                                             unsigned short* __restrict__ sq) {
  __shared__ float red[256];
  long row = blockIdx.x;
  const float* srow = s + row * (long)NS;
  float kv[4], qv[4];
#pragma unroll
  for (int j = 0; j < 4; ++j) kv[j] = srow[threadIdx.x + j * 256];
#pragma unroll
  for (int j = 0; j < 4; ++j) qv[j] = srow[2048 + threadIdx.x + j * 256];

  // softmax(k) + sigmoid(k)
  float mx = blockMax(fmaxf(fmaxf(kv[0], kv[1]), fmaxf(kv[2], kv[3])), red);
  float e[4], ps = 0.f;
#pragma unroll
  for (int j = 0; j < 4; ++j) { e[j] = __expf(kv[j] - mx); ps += e[j]; }
  float inv = 1.f / blockSum(ps, red);
#pragma unroll
  for (int j = 0; j < 4; ++j) {
    long c = row * (long)DIM + threadIdx.x + j * 256;
    sk[c] = f2bf(e[j] * inv);
    gk[c] = f2bf(1.f / (1.f + __expf(-kv[j])));
  }

  // softmax(q)
  mx = blockMax(fmaxf(fmaxf(qv[0], qv[1]), fmaxf(qv[2], qv[3])), red);
  ps = 0.f;
#pragma unroll
  for (int j = 0; j < 4; ++j) { e[j] = __expf(qv[j] - mx); ps += e[j]; }
  inv = 1.f / blockSum(ps, red);
#pragma unroll
  for (int j = 0; j < 4; ++j)
    sq[row * (long)DIM + threadIdx.x + j * 256] = f2bf(e[j] * inv);
}

// ---------------------------------------------------------------- delta GEMM
// W_new[o][h] = bf16( W_fast[o][h] + (1/B) * sum_b r[b][o]*gk[b][h] )
__global__ __launch_bounds__(256) void k_delta_wnew(
    const unsigned short* __restrict__ r,   // BATCH x DIM bf16
    const unsigned short* __restrict__ gk,  // BATCH x DIM bf16
    const float* __restrict__ Wfast,        // DIM x DIM f32
    unsigned short* __restrict__ Wnew)      // DIM x DIM bf16
{
  __shared__ unsigned short lA[TM * LSTR];  // [o_local][b_local] (transposed)
  __shared__ unsigned short lB[TN * LSTR];  // [h_local][b_local] (transposed)

  const int tid  = threadIdx.x;
  const int wave = tid >> 5;
  const int lane = tid & 31;
  const int half = lane >> 4;
  const int l16  = lane & 15;

  const long oBase = (long)blockIdx.y * TM;
  const long hBase = (long)blockIdx.x * TN;

  v8f acc[8] = {};

  const int kb = tid >> 3;          // 0..31  : batch row within K-tile
  const int c0 = (tid & 7) * 16;    // 0..112 : 16-wide column chunk

  for (int k0 = 0; k0 < BATCH; k0 += TK) {
    const unsigned short* srcR = r  + (long)(k0 + kb) * DIM + oBase + c0;
    const unsigned short* srcG = gk + (long)(k0 + kb) * DIM + hBase + c0;
    if (k0 + TK < BATCH) {            // gfx1250 global_prefetch_b8
      __builtin_prefetch(srcR + (long)TK * DIM, 0, 1);
      __builtin_prefetch(srcG + (long)TK * DIM, 0, 1);
    }
    __syncthreads();
    {
      U16x8 a0, a1; a0.q = *(const uint4*)srcR; a1.q = *(const uint4*)(srcR + 8);
#pragma unroll
      for (int j = 0; j < 8; ++j) lA[(c0 + j) * LSTR + kb] = a0.h[j];
#pragma unroll
      for (int j = 0; j < 8; ++j) lA[(c0 + 8 + j) * LSTR + kb] = a1.h[j];
      U16x8 b0, b1; b0.q = *(const uint4*)srcG; b1.q = *(const uint4*)(srcG + 8);
#pragma unroll
      for (int j = 0; j < 8; ++j) lB[(c0 + j) * LSTR + kb] = b0.h[j];
#pragma unroll
      for (int j = 0; j < 8; ++j) lB[(c0 + 8 + j) * LSTR + kb] = b1.h[j];
    }
    __syncthreads();

    BF16x16 bf;
    {
      const unsigned short* p = &lB[(wave * 16 + l16) * LSTR + half * 8];
      uint4 q0 = *(const uint4*)(p);
      uint4 q1 = *(const uint4*)(p + 16);
      bf.u[0]=q0.x; bf.u[1]=q0.y; bf.u[2]=q0.z; bf.u[3]=q0.w;
      bf.u[4]=q1.x; bf.u[5]=q1.y; bf.u[6]=q1.z; bf.u[7]=q1.w;
    }
    BF16x16 af[8];
#pragma unroll
    for (int mt = 0; mt < 8; ++mt) {
      const unsigned short* p = &lA[(mt * 16 + l16) * LSTR + half * 8];
      uint4 q0 = *(const uint4*)(p);
      uint4 q1 = *(const uint4*)(p + 16);
      af[mt].u[0]=q0.x; af[mt].u[1]=q0.y; af[mt].u[2]=q0.z; af[mt].u[3]=q0.w;
      af[mt].u[4]=q1.x; af[mt].u[5]=q1.y; af[mt].u[6]=q1.z; af[mt].u[7]=q1.w;
    }
#pragma unroll
    for (int mt = 0; mt < 8; ++mt)
      acc[mt] = __builtin_amdgcn_wmma_f32_16x16x32_bf16(
          false, af[mt].v, false, bf.v, (short)0, acc[mt], false, false);
  }

  const long h = hBase + wave * 16 + l16;
  const float invB = 1.0f / (float)BATCH;
#pragma unroll
  for (int mt = 0; mt < 8; ++mt) {
#pragma unroll
    for (int i = 0; i < 8; ++i) {
      long o = oBase + mt * 16 + half * 8 + i;
      Wnew[o * (long)DIM + h] = f2bf(Wfast[o * (long)DIM + h] + acc[mt][i] * invB);
    }
  }
}

// ---------------------------------------------------------------- launcher
extern "C" void kernel_launch(void* const* d_in, const int* in_sizes, int n_in,
                              void* d_out, int out_size, void* d_ws, size_t ws_size,
                              hipStream_t stream) {
  const float* x       = (const float*)d_in[0];
  const float* Wslow_w = (const float*)d_in[1];   // (3073, 1024)
  const float* Wslow_b = (const float*)d_in[2];   // (3073,)
  const float* Wfast_w = (const float*)d_in[3];   // (1024, 1024)
  const float* Wfast_b = (const float*)d_in[4];   // (1024,)
  float* out = (float*)d_out;

  char* p = (char*)d_ws;
  auto alloc = [&](size_t bytes) {
    char* r = p; p += (bytes + 255) & ~(size_t)255; return r;
  };
  unsigned short* xb   = (unsigned short*)alloc((size_t)BATCH * DIM * 2);
  unsigned short* Wsb  = (unsigned short*)alloc((size_t)NS * DIM * 2);
  unsigned short* Wfb  = (unsigned short*)alloc((size_t)DIM * DIM * 2);
  float*          s    = (float*)         alloc((size_t)BATCH * NS * 4);
  float*          lr   = (float*)         alloc((size_t)BATCH * 4);
  unsigned short* sk   = (unsigned short*)alloc((size_t)BATCH * DIM * 2);
  unsigned short* gk   = (unsigned short*)alloc((size_t)BATCH * DIM * 2);
  unsigned short* sq   = (unsigned short*)alloc((size_t)BATCH * DIM * 2);
  unsigned short* rbuf = (unsigned short*)alloc((size_t)BATCH * DIM * 2);
  unsigned short* Wnb  = (unsigned short*)alloc((size_t)DIM * DIM * 2);

  // bf16 staging
  k_f32_to_bf16<<<2048, 256, 0, stream>>>(x, xb, (long)BATCH * DIM);
  k_f32_to_bf16<<<1024, 256, 0, stream>>>(Wslow_w, Wsb, (long)NS * DIM);
  k_f32_to_bf16<<<512, 256, 0, stream>>>(Wfast_w, Wfb, (long)DIM * DIM);

  // GEMM1: s = x @ Wslow[:3072]^T + b[:3072]
  k_wmma_gemm<<<dim3(NS / TN, BATCH / TM), 256, 0, stream>>>(
      xb, DIM, Wsb, DIM, Wslow_b, DIM, /*mode=*/0,
      s, NS, nullptr, nullptr, nullptr, 0);

  // lr = sigmoid(x @ Wslow[3072] + b[3072])
  k_lr<<<BATCH, 256, 0, stream>>>(x, Wslow_w + (size_t)NS * DIM, Wslow_b + NS, lr);

  // row activations: softmax(k), sigmoid(k), softmax(q)
  k_act<<<BATCH, 256, 0, stream>>>(s, sk, gk, sq);

  // GEMM2 fused: r = lr * (v - (softmax(k) @ Wfast^T + b_fast)),  v = s[:,1024:2048]
  k_wmma_gemm<<<dim3(DIM / TN, BATCH / TM), 256, 0, stream>>>(
      sk, DIM, Wfb, DIM, Wfast_b, DIM, /*mode=*/1,
      nullptr, 0, rbuf, lr, s + DIM, NS);

  // W_new = Wfast + (r^T @ gk)/B  (bf16, ready as GEMM4 B-operand)
  k_delta_wnew<<<dim3(DIM / TN, DIM / TM), 256, 0, stream>>>(rbuf, gk, Wfast_w, Wnb);

  // GEMM4: out = softmax(q) @ W_new^T + b_fast
  k_wmma_gemm<<<dim3(DIM / TN, BATCH / TM), 256, 0, stream>>>(
      sq, DIM, Wnb, DIM, Wfast_b, DIM, /*mode=*/0,
      out, DIM, nullptr, nullptr, nullptr, 0);
}
